// InteractionPruning_72378788872881
// MI455X (gfx1250) — matrix operand order repaired and checked by reading, not compiled, not run on hardware
//
#include <hip/hip_runtime.h>
#include <hip/hip_bf16.h>

// Problem constants (from reference)
#define BB   1024
#define FF   32
#define DD   128
#define NPAIR ((FF * (FF - 1)) / 2)   // 496
#define ZETA 1.1f
#define GAMMA (-0.1f)

// Tiling
#define BT   64            // batch rows per block
#define ZSTR 136           // LDS row stride (bf16): 128 data + 4-DWORD TDM pad (272B = 17*16)
#define FSTR 136
#define MSTR 132           // LDS row stride (f32) for M

typedef __attribute__((ext_vector_type(16))) __bf16 v16bf;
typedef __attribute__((ext_vector_type(8)))  __bf16 v8bf;
typedef __attribute__((ext_vector_type(8)))  float  v8f;
typedef __attribute__((ext_vector_type(4)))  unsigned int v4u;
typedef __attribute__((ext_vector_type(8)))  int v8i;
typedef __attribute__((ext_vector_type(4)))  int v4i;
typedef __attribute__((ext_vector_type(4)))  unsigned short v4us;

__device__ __forceinline__ unsigned short f2bf(float f) {
  // round-to-nearest-even f32 -> bf16
  unsigned int u = __builtin_bit_cast(unsigned int, f);
  u += 0x7FFFu + ((u >> 16) & 1u);
  return (unsigned short)(u >> 16);
}

__device__ __forceinline__ unsigned short gate_bf16(float x) {
  float s = 1.0f / (1.0f + __expf(-x));
  float z = s * (ZETA - GAMMA) + GAMMA;
  return f2bf(fminf(fmaxf(z, 0.0f), 1.0f));
}

// A fragment (16-bit A 16x32): two v8bf at +0 / +8 from the lane's K base.
__device__ __forceinline__ v16bf load_frag_a(const unsigned short* p) {
  v8bf x = *(const v8bf*)(p);
  v8bf y = *(const v8bf*)(p + 8);
  return __builtin_shufflevector(x, y, 0,1,2,3,4,5,6,7,8,9,10,11,12,13,14,15);
}
// B fragment (16-bit B 32x16): 16 contiguous bf16 per lane.
__device__ __forceinline__ v16bf load_frag_b(const unsigned short* p) {
  v8bf x = *(const v8bf*)(p);
  v8bf y = *(const v8bf*)(p + 8);
  return __builtin_shufflevector(x, y, 0,1,2,3,4,5,6,7,8,9,10,11,12,13,14,15);
}

// ---- Tensor Data Mover: 2D bf16 tile, global -> LDS, with HW row padding -----
// D# per CDNA5 ISA ch.8: group0 {count, lds_addr, global_addr, type=2},
// group1 {data_size=2B, pad 4 DWORDs after every 64 DWORDs (256B row -> 272B LDS
// stride), tensor dims, tile dims, dim0 stride}. Remaining groups zero (2D tensor).
// This toolchain exposes the 6-arg builtin: (v4u, v8i, v4i, v4i, v8i, i32 cpol).
__device__ __forceinline__ void tdm_load_tile_2d(unsigned lds_addr,
                                                 const unsigned short* gptr,
                                                 unsigned tensor_d0, unsigned tensor_d1,
                                                 unsigned tile_d0,   unsigned tile_d1,
                                                 unsigned long long d0_stride /*elems*/) {
  unsigned long long ga = (unsigned long long)(uintptr_t)gptr;
  v4u g0;
  g0[0] = 1u;                                                   // count=1, user descriptor
  g0[1] = lds_addr;                                             // LDS byte address
  g0[2] = (unsigned)(ga & 0xFFFFFFFFu);                         // global_addr[31:0]
  g0[3] = (unsigned)((ga >> 32) & 0x01FFFFFFu) | (2u << 30);    // global_addr[56:32], type=2
  v8i g1;
  g1[0] = (int)((1u << 16) | (1u << 20) | (5u << 22) | (3u << 25)); // 2B elems, pad_en, ivl=64DW, amt=4DW
  g1[1] = (int)((tensor_d0 & 0xFFFFu) << 16);                   // tensor_dim0[15:0]
  g1[2] = (int)((tensor_d0 >> 16) | ((tensor_d1 & 0xFFFFu) << 16));
  g1[3] = (int)((tensor_d1 >> 16) | ((tile_d0 & 0xFFFFu) << 16));
  g1[4] = (int)(tile_d1 & 0xFFFFu);                             // tile_dim1, tile_dim2=0
  g1[5] = (int)(d0_stride & 0xFFFFFFFFu);                       // tensor_dim0_stride[31:0]
  g1[6] = (int)((d0_stride >> 32) & 0xFFFFu);                   // stride[47:32]; dim1_stride=0 (2D)
  g1[7] = 0;
  v4i z4 = {0, 0, 0, 0};
  v8i z8 = {0, 0, 0, 0, 0, 0, 0, 0};
  __builtin_amdgcn_tensor_load_to_lds(g0, g1, z4, z4, z8, 0);
}

// ---------------- prep kernel 1: feature f32 -> bf16 scratch ----------------
__global__ __launch_bounds__(256)
void prep_feature_bf16(const float* __restrict__ f, unsigned short* __restrict__ o) {
  size_t idx = ((size_t)blockIdx.x * 256 + threadIdx.x) * 4;
  float4 v = *(const float4*)(f + idx);
  v4us r = { f2bf(v.x), f2bf(v.y), f2bf(v.z), f2bf(v.w) };
  *(v4us*)(o + idx) = r;
}

// ------- prep kernel 2: Z_p = clip(sigmoid(matrix[i,j])*1.2-0.1,0,1) bf16 ------
__global__ __launch_bounds__(256)
void prep_gate_bf16(const float* __restrict__ matrix, unsigned short* __restrict__ zout) {
  int rem = blockIdx.y, i = 0;
  while (rem >= (FF - 1 - i)) { rem -= (FF - 1 - i); ++i; }
  const int j = i + 1 + rem;
  size_t off = ((size_t)blockIdx.x * 256 + threadIdx.x) * 4;    // within [0, 16384)
  const float* mp = matrix + (size_t)(i * FF + j) * DD * DD + off;
  float4 x = *(const float4*)mp;
  v4us r = { gate_bf16(x.x), gate_bf16(x.y), gate_bf16(x.z), gate_bf16(x.w) };
  *(v4us*)(zout + (size_t)blockIdx.y * DD * DD + off) = r;
}

// ---------------- main kernel: TDM staging + WMMA + dot epilogue ----------------
__global__ __launch_bounds__(256)
void interaction_pruning_pair_kernel(const float* __restrict__ feature,
                                     const unsigned short* __restrict__ wsf,   // [B,F,D] bf16
                                     const unsigned short* __restrict__ wsz,   // [496,D,D] bf16
                                     float* __restrict__ out) {
  // LDS: Zs (34816B) + FJ (17408B) = 52224B; Ms (33792B) overlays Zs after WMMA.
  __shared__ __align__(32) unsigned char smem[52224];
  unsigned short* Zs = (unsigned short*)smem;                 // 128 x ZSTR bf16
  unsigned short* FJ = (unsigned short*)(smem + 34816);       // BT  x FSTR bf16
  float*          Ms = (float*)smem;                          // BT  x MSTR f32 (overlay)

  const int t = threadIdx.x;
  const int pair = blockIdx.y;
  int rem = pair, i = 0;
  while (rem >= (FF - 1 - i)) { rem -= (FF - 1 - i); ++i; }
  const int j = i + 1 + rem;
  const int bbase = blockIdx.x * BT;

  // ---- Phase A: Tensor Data Mover stages both bf16 tiles into padded LDS ----
  if (t < 32) {   // one wave issues the DMA (TDM ignores EXEC; uniform in wave 0)
    unsigned lds_base = (unsigned)(uintptr_t)smem;            // LDS byte offset
    // Z_pair: 128x128 tile, rows contiguous (stride 128 elems)
    tdm_load_tile_2d(lds_base, wsz + (size_t)pair * DD * DD,
                     DD, DD, DD, DD, DD);
    // f_j tile: 64 batch rows, row stride F*D elems
    tdm_load_tile_2d(lds_base + 34816u, wsf + ((size_t)bbase * FF + j) * DD,
                     DD, BT, DD, BT, (unsigned long long)FF * DD);
  }
  // warm caches with the f32 f_i rows the epilogue reads
  __builtin_prefetch(&feature[((size_t)(bbase + (t >> 2)) * FF + i) * DD + 32 * (t & 3)], 0, 3);
  __builtin_amdgcn_s_wait_tensorcnt(0);
  __syncthreads();

  // ---- Phase B: M[b,d] = sum_e Z[d,e] * f_j[b,e] via bf16 WMMA ----
  const int wave  = t >> 5;                 // 8 wave32
  const int lane  = t & 31;
  const int l16   = lane & 15;
  const int hi    = lane >> 4;              // K-split half
  const int strip = wave & 3;               // 16-row M strip
  const int nbase = (wave >> 2) * 4;        // 4 N-tiles per wave

  // Hoist all lane base addresses: per-k offsets become instruction immediates.
  const unsigned short* arow  = FJ + (16 * strip + l16) * FSTR + 16 * hi;
  const unsigned short* brow0 = Zs + (16 * (nbase + 0) + l16) * ZSTR + 16 * hi;
  const unsigned short* brow1 = Zs + (16 * (nbase + 1) + l16) * ZSTR + 16 * hi;
  const unsigned short* brow2 = Zs + (16 * (nbase + 2) + l16) * ZSTR + 16 * hi;
  const unsigned short* brow3 = Zs + (16 * (nbase + 3) + l16) * ZSTR + 16 * hi;

  v8f acc0 = {}, acc1 = {}, acc2 = {}, acc3 = {};

  // Software pipeline: fragments for step k+1 load into fresh registers while
  // step k's WMMAs issue, so ds_load latency overlaps the matrix pipe.
  v16bf a  = load_frag_a(arow);
  v16bf b0 = load_frag_b(brow0);
  v16bf b1 = load_frag_b(brow1);
  v16bf b2 = load_frag_b(brow2);
  v16bf b3 = load_frag_b(brow3);

  #pragma unroll
  for (int k = 0; k < 4; ++k) {
    v16bf an, bn0, bn1, bn2, bn3;
    if (k < 3) {
      const int k0 = 32 * (k + 1);
      an  = load_frag_a(arow + k0);
      bn0 = load_frag_b(brow0 + k0);
      bn1 = load_frag_b(brow1 + k0);
      bn2 = load_frag_b(brow2 + k0);
      bn3 = load_frag_b(brow3 + k0);
    }
    acc0 = __builtin_amdgcn_wmma_f32_16x16x32_bf16(false, a, false, b0, (short)0, acc0, false, false);
    acc1 = __builtin_amdgcn_wmma_f32_16x16x32_bf16(false, a, false, b1, (short)0, acc1, false, false);
    acc2 = __builtin_amdgcn_wmma_f32_16x16x32_bf16(false, a, false, b2, (short)0, acc2, false, false);
    acc3 = __builtin_amdgcn_wmma_f32_16x16x32_bf16(false, a, false, b3, (short)0, acc3, false, false);
    if (k < 3) { a = an; b0 = bn0; b1 = bn1; b2 = bn2; b3 = bn3; }
  }

  __syncthreads();   // all LDS reads done; safe to overlay M on Z region

  {
    const int mrow = 16 * strip + 8 * hi;
    float* mbase = Ms + mrow * MSTR + l16;
    #pragma unroll
    for (int r = 0; r < 8; ++r) {
      mbase[(r * MSTR) + 16 * (nbase + 0)] = acc0[r];
      mbase[(r * MSTR) + 16 * (nbase + 1)] = acc1[r];
      mbase[(r * MSTR) + 16 * (nbase + 2)] = acc2[r];
      mbase[(r * MSTR) + 16 * (nbase + 3)] = acc3[r];
    }
  }
  __syncthreads();

  // ---- Phase C: out[b,i,j] = f_i[b,:] . M[b,:]  (f32 f_i for accuracy) ----
  const int row = t >> 2;
  const int q   = t & 3;
  const int b   = bbase + row;
  const float* fi = feature + ((size_t)b * FF + i) * DD + 32 * q;
  const float* mr = Ms + row * MSTR + 32 * q;
  float sum = 0.0f;
  #pragma unroll
  for (int d = 0; d < 32; ++d) sum += fi[d] * mr[d];
  sum += __shfl_xor(sum, 1, 32);
  sum += __shfl_xor(sum, 2, 32);
  if (q == 0) out[(size_t)b * FF * FF + i * FF + j] = sum;
}

extern "C" void kernel_launch(void* const* d_in, const int* in_sizes, int n_in,
                              void* d_out, int out_size, void* d_ws, size_t ws_size,
                              hipStream_t stream) {
  const float* feature = (const float*)d_in[0];   // [B, F, D] f32
  const float* matrix  = (const float*)d_in[1];   // [F, F, D, D] f32
  float* out = (float*)d_out;                     // [B, F, F] f32

  // scratch layout: [0, 8MB) feature bf16; [8MB, 23.5MB) gated pair matrices bf16
  unsigned short* wsf = (unsigned short*)d_ws;
  unsigned short* wsz = wsf + (size_t)BB * FF * DD;

  // zero the diagonal + lower triangle (reference multiplies by strict-upper mask)
  (void)hipMemsetAsync(out, 0, (size_t)out_size * sizeof(float), stream);

  // one-time precompute per call: bf16 feature + bf16 hard-concrete gates
  prep_feature_bf16<<<(BB * FF * DD) / (256 * 4), 256, 0, stream>>>(feature, wsf);
  prep_gate_bf16<<<dim3((DD * DD) / (256 * 4), NPAIR), 256, 0, stream>>>(matrix, wsz);

  dim3 grid(BB / BT, NPAIR);                      // 16 batch tiles x 496 pairs
  interaction_pruning_pair_kernel<<<grid, 256, 0, stream>>>(feature, wsf, wsz, out);
}
